// SDCN_42185168781981
// MI455X (gfx1250) — matrix operand order repaired and checked by reading, not compiled, not run on hardware
//
#include <hip/hip_runtime.h>
#include <hip/hip_bf16.h>
#include <math.h>

// ---------------------------------------------------------------------------
// Types for CDNA5 WMMA (wave32): D(16x16 f32) = A(16x32 bf16) x B(32x16 bf16) + C
// ---------------------------------------------------------------------------
typedef __attribute__((ext_vector_type(16))) __bf16 v16bf;
typedef __attribute__((ext_vector_type(8)))  __bf16 v8bf;
typedef __attribute__((ext_vector_type(8)))  float  v8f;

#define BN_EPS 1e-5f
#define GEMM_MT 2   // M tiles per wave (32 rows)
#define GEMM_NT 2   // N tiles per wave (32 cols)

// ---------------------------------------------------------------------------
// GEMM: C[M][ldc] = A[M][Kp](bf16) * Bt[Nc][Kp]^T (bf16) + bias   (f32 accum)
// One wave computes a 32x32 tile (2x2 x 16x16 WMMA): per K-step this costs
// 4 A-loads + 4 B-loads (b128) for 4 WMMAs -- balanced register-level reuse.
// Kp % 32 == 0, M % 32 == 0. B row indices are clamped (not selected):
// garbage columns only affect output columns the store guard skips.
// ---------------------------------------------------------------------------
__global__ void sdcn_gemm_bf16_wmma(const __bf16* __restrict__ A,
                                    const __bf16* __restrict__ Bt,
                                    const float*  __restrict__ bias,
                                    float*        __restrict__ C,
                                    int M, int Nc, int Kp, int ldc,
                                    int tilesN2, long totalTiles)
{
    const long waveId = (long)blockIdx.x * (blockDim.x >> 5) + (threadIdx.x >> 5);
    if (waveId >= totalTiles) return;
    const int tileM2 = (int)(waveId / tilesN2);
    const int tileN2 = (int)(waveId % tilesN2);
    const int lane   = threadIdx.x & 31;

    // A layout (16-bit A 16x32): lanes 0-15 row M=lane, k in {0..7,16..23};
    // lanes 16-31 same rows, k in {8..15,24..31}.
    const int aKoff = (lane >> 4) * 8;
    const __bf16* arow[GEMM_MT];
    #pragma unroll
    for (int i = 0; i < GEMM_MT; ++i) {
        const int mi = tileM2 * (16 * GEMM_MT) + i * 16 + (lane & 15);
        arow[i] = A + (size_t)mi * Kp + aKoff;
    }

    // B layout (32x16): lanes 0-15 col N=lane, k in {0..7,16..23}; lanes 16-31 k in {8..15,24..31}.
    const int bKoff = (lane >> 4) * 8;
    int ncol[GEMM_NT];
    const __bf16* brow[GEMM_NT];
    #pragma unroll
    for (int j = 0; j < GEMM_NT; ++j) {
        ncol[j] = tileN2 * (16 * GEMM_NT) + j * 16 + (lane & 15);
        const int nld = min(ncol[j], Nc - 1);          // clamp: keep loads in-bounds
        brow[j] = Bt + (size_t)nld * Kp + bKoff;
    }

    v8f acc[GEMM_MT][GEMM_NT];
    #pragma unroll
    for (int i = 0; i < GEMM_MT; ++i)
        #pragma unroll
        for (int j = 0; j < GEMM_NT; ++j)
            #pragma unroll
            for (int r = 0; r < 8; ++r) acc[i][j][r] = 0.0f;

    for (int k0 = 0; k0 < Kp; k0 += 32) {
        v16bf av[GEMM_MT];
        #pragma unroll
        for (int i = 0; i < GEMM_MT; ++i) {
            __builtin_prefetch(arow[i] + k0 + 128, 0, 3);   // pull A into near cache
            const v8bf a0 = *(const v8bf*)(arow[i] + k0);
            const v8bf a1 = *(const v8bf*)(arow[i] + k0 + 16);
            #pragma unroll
            for (int e = 0; e < 8; ++e) { av[i][e] = a0[e]; av[i][e + 8] = a1[e]; }
        }
        #pragma unroll
        for (int j = 0; j < GEMM_NT; ++j) {
            const v8bf b0 = *(const v8bf*)(brow[j] + k0);  // B is L2-resident
            const v8bf b1 = *(const v8bf*)(brow[j] + k0 + 16);
            v16bf bv;
            #pragma unroll
            for (int e = 0; e < 8; ++e) { bv[e] = b0[e]; bv[e + 8] = b1[e]; }
            #pragma unroll
            for (int i = 0; i < GEMM_MT; ++i) {
                acc[i][j] = __builtin_amdgcn_wmma_f32_16x16x32_bf16(
                                /*neg_a=*/false, av[i], /*neg_b=*/false, bv,
                                /*c_mod=*/(short)0, acc[i][j],
                                /*reuse_a=*/false, /*reuse_b=*/false);
            }
        }
    }

    // Store: lanes 0-15 hold rows +0..7 of each tile, lanes 16-31 rows +8..15.
    #pragma unroll
    for (int i = 0; i < GEMM_MT; ++i) {
        const int mrow = tileM2 * (16 * GEMM_MT) + i * 16 + (lane >> 4) * 8;
        #pragma unroll
        for (int j = 0; j < GEMM_NT; ++j) {
            if (ncol[j] < Nc) {
                const float bb = bias ? bias[ncol[j]] : 0.0f;
                #pragma unroll
                for (int r = 0; r < 8; ++r)
                    C[(size_t)(mrow + r) * ldc + ncol[j]] = acc[i][j][r] + bb;
            }
        }
    }
}

// ---------------------------------------------------------------------------
// Weight conversion: out[n][k] (bf16, k padded to Kp) from f32 source.
// transposed==0: in is [dout][din] row-major (nn.Linear weight, used as h@W^T)
// transposed==1: in is [din][dout] row-major (GNN weight, used as h@W)
// ---------------------------------------------------------------------------
__global__ void sdcn_cvt_weight(const float* __restrict__ in, __bf16* __restrict__ out,
                                int dout, int din, int Kp, int transposed)
{
    const size_t idx = (size_t)blockIdx.x * blockDim.x + threadIdx.x;
    const size_t tot = (size_t)dout * Kp;
    if (idx >= tot) return;
    const int n = (int)(idx / Kp);
    const int k = (int)(idx % Kp);
    float v = 0.0f;
    if (k < din) v = transposed ? in[(size_t)k * dout + n] : in[(size_t)n * din + k];
    out[idx] = (__bf16)v;
}

// Activation f32 -> bf16 with K padding.
__global__ void sdcn_cvt_act(const float* __restrict__ in, __bf16* __restrict__ out,
                             int M, int K, int Kp)
{
    const size_t idx = (size_t)blockIdx.x * blockDim.x + threadIdx.x;
    const size_t tot = (size_t)M * Kp;
    if (idx >= tot) return;
    const int m = (int)(idx / Kp);
    const int k = (int)(idx % Kp);
    out[idx] = (__bf16)((k < K) ? in[(size_t)m * K + k] : 0.0f);
}

// ---------------------------------------------------------------------------
// BatchNorm1d (training mode, biased variance): per-column stats over M rows.
// ---------------------------------------------------------------------------
__global__ void sdcn_bn_stats(const float* __restrict__ H, int M, int Nc,
                              float* __restrict__ mean, float* __restrict__ rstd)
{
    const int c = blockIdx.x;
    __shared__ float s0[256], s1[256];
    float sum = 0.0f, sq = 0.0f;
    for (int r = threadIdx.x; r < M; r += blockDim.x) {
        const float v = H[(size_t)r * Nc + c];
        sum += v; sq += v * v;
    }
    s0[threadIdx.x] = sum; s1[threadIdx.x] = sq;
    __syncthreads();
    for (int s = 128; s > 0; s >>= 1) {
        if ((int)threadIdx.x < s) {
            s0[threadIdx.x] += s0[threadIdx.x + s];
            s1[threadIdx.x] += s1[threadIdx.x + s];
        }
        __syncthreads();
    }
    if (threadIdx.x == 0) {
        const float mu  = s0[0] / (float)M;
        const float var = s1[0] / (float)M - mu * mu;
        mean[c] = mu;
        rstd[c] = rsqrtf(var + BN_EPS);
    }
}

// Apply BN (+optional ReLU); write bf16 (padded) and optionally f32.
__global__ void sdcn_bn_apply(const float* __restrict__ H,
                              const float* __restrict__ mean,
                              const float* __restrict__ rstd,
                              const float* __restrict__ g,
                              const float* __restrict__ b,
                              float* __restrict__ outF,
                              __bf16* __restrict__ outB,
                              int M, int Nc, int Kp, int relu)
{
    const size_t idx = (size_t)blockIdx.x * blockDim.x + threadIdx.x;
    const size_t tot = (size_t)M * Kp;
    if (idx >= tot) return;
    const int m = (int)(idx / Kp);
    const int k = (int)(idx % Kp);
    float v = 0.0f;
    if (k < Nc) {
        v = (H[(size_t)m * Nc + k] - mean[k]) * rstd[k] * g[k] + b[k];
        if (relu) v = fmaxf(v, 0.0f);
        if (outF) outF[(size_t)m * Nc + k] = v;
    }
    outB[idx] = (__bf16)v;
}

// ---------------------------------------------------------------------------
// SpMM scatter: out[dst[e], :] += w[e] * h[src[e], :]    (block per edge)
// ---------------------------------------------------------------------------
__global__ void sdcn_spmm_scatter(const int* __restrict__ src, const int* __restrict__ dst,
                                  const float* __restrict__ w, const float* __restrict__ h,
                                  float* __restrict__ out, int E, int Nc)
{
    const int e = blockIdx.x;
    if (e >= E) return;
    const int s = src[e], d = dst[e];
    const float wt = w[e];
    const float* hr = h + (size_t)s * Nc;
    float* orow = out + (size_t)d * Nc;
    for (int c = threadIdx.x; c < Nc; c += blockDim.x)
        __hip_atomic_fetch_add(&orow[c], hr[c] * wt, __ATOMIC_RELAXED,
                               __HIP_MEMORY_SCOPE_AGENT);
}

// relu(spmm_out) mixed with sigma*t, repacked to bf16 (t has the same Kp stride)
__global__ void sdcn_combine_relu(const float* __restrict__ hs,
                                  const __bf16* __restrict__ t,
                                  __bf16* __restrict__ out,
                                  int M, int Nc, int Kp, float sgm, int relu)
{
    const size_t idx = (size_t)blockIdx.x * blockDim.x + threadIdx.x;
    const size_t tot = (size_t)M * Kp;
    if (idx >= tot) return;
    const int m = (int)(idx / Kp);
    const int k = (int)(idx % Kp);
    float v = 0.0f;
    if (k < Nc) {
        float hv = hs[(size_t)m * Nc + k];
        if (relu) hv = fmaxf(hv, 0.0f);
        v = (1.0f - sgm) * hv + sgm * (float)t[idx];
    }
    out[idx] = (__bf16)v;
}

// Row softmax over C columns (C small, here 10).
__global__ void sdcn_softmax_rows(const float* __restrict__ in, float* __restrict__ out,
                                  int M, int C)
{
    const int m = blockIdx.x * blockDim.x + threadIdx.x;
    if (m >= M) return;
    const float* r = in + (size_t)m * C;
    float mx = -3.4e38f;
    for (int c = 0; c < C; ++c) mx = fmaxf(mx, r[c]);
    float s = 0.0f;
    for (int c = 0; c < C; ++c) s += expf(r[c] - mx);
    const float inv = 1.0f / s;
    for (int c = 0; c < C; ++c) out[(size_t)m * C + c] = expf(r[c] - mx) * inv;
}

// Student-t cluster assignment (V=1 -> exponent 1): q = 1/(1+||z-mu||^2), row-normalized
__global__ void sdcn_student_q(const float* __restrict__ z, const float* __restrict__ cl,
                               float* __restrict__ q, int M)
{
    const int m = blockIdx.x * blockDim.x + threadIdx.x;
    if (m >= M) return;
    float qa[10];
    float s = 0.0f;
    const float* zr = z + (size_t)m * 32;
    for (int c = 0; c < 10; ++c) {
        float d = 0.0f;
        for (int k = 0; k < 32; ++k) {
            const float t = zr[k] - cl[c * 32 + k];
            d += t * t;
        }
        const float v = 1.0f / (1.0f + d);
        qa[c] = v; s += v;
    }
    const float inv = 1.0f / s;
    for (int c = 0; c < 10; ++c) q[(size_t)m * 10 + c] = qa[c] * inv;
}

// Decoder heads: 0 = MeanAct clip(exp, 1e-5, 1e6)
//                1 = DispAct clip(softplus, 1e-4, 1e4)
//                2 = sigmoid
__global__ void sdcn_heads(const float* __restrict__ in, float* __restrict__ out,
                           size_t n, int mode)
{
    const size_t i = (size_t)blockIdx.x * blockDim.x + threadIdx.x;
    if (i >= n) return;
    const float x = in[i];
    float r;
    if (mode == 0) {
        r = fminf(fmaxf(expf(x), 1e-5f), 1e6f);
    } else if (mode == 1) {
        const float sp = fmaxf(x, 0.0f) + log1pf(expf(-fabsf(x)));
        r = fminf(fmaxf(sp, 1e-4f), 1e4f);
    } else {
        r = 1.0f / (1.0f + expf(-x));
    }
    out[i] = r;
}

// ---------------------------------------------------------------------------
// Host-side launch
// ---------------------------------------------------------------------------
static inline int ceil_div_i(size_t a, int b) { return (int)((a + b - 1) / b); }

extern "C" void kernel_launch(void* const* d_in, const int* in_sizes, int n_in,
                              void* d_out, int out_size, void* d_ws, size_t ws_size,
                              hipStream_t stream)
{
    (void)in_sizes; (void)n_in; (void)out_size; (void)ws_size;

    const int N = 20000, E = 320000;
    const float SIGMA = 0.5f;

    const float* x    = (const float*)d_in[0];
    const int*   esrc = (const int*)d_in[1];
    const int*   edst = (const int*)d_in[2];
    const float* ew   = (const float*)d_in[3];
    const float* cluster = (const float*)d_in[55];

    float* out = (float*)d_out;
    float* o_xbar = out;
    float* o_q    = o_xbar + (size_t)N * 2000;
    float* o_pred = o_q    + (size_t)N * 10;
    float* o_z3   = o_pred + (size_t)N * 10;
    float* o_mean = o_z3   + (size_t)N * 32;
    float* o_disp = o_mean + (size_t)N * 2000;
    float* o_pi   = o_disp + (size_t)N * 2000;

    // ---- workspace bump allocator (256B aligned) ----
    size_t off = 0;
    char* wsb = (char*)d_ws;
    auto wsAlloc = [&](size_t bytes) -> void* {
        void* p = wsb + off;
        off += (bytes + 255) & ~(size_t)255;
        return p;
    };

    // ---- weight conversions to bf16 Bt[N][Kp] ----
    struct WDesc { int idx; int dout; int din; int transposed; };
    const WDesc wd[20] = {
        {4, 1024, 2000, 0}, {6, 512, 1024, 0}, {8, 256, 512, 0},
        {10, 128, 256, 0},  {12, 64, 128, 0},  {14, 32, 64, 0},
        {16, 256, 32, 0},   {18, 512, 256, 0}, {20, 1024, 512, 0},
        {22, 2000, 1024, 0},{24, 2000, 1024, 0},{26, 2000, 1024, 0},
        {28, 2000, 1024, 0},
        {48, 1024, 2000, 1},{49, 512, 1024, 1},{50, 256, 512, 1},
        {51, 128, 256, 1},  {52, 64, 128, 1},  {53, 32, 64, 1},
        {54, 10, 32, 1},
    };
    __bf16* wbf[20];
    for (int i = 0; i < 20; ++i) {
        const int Kp = (wd[i].din + 31) & ~31;
        const size_t tot = (size_t)wd[i].dout * Kp;
        wbf[i] = (__bf16*)wsAlloc(tot * sizeof(__bf16));
        sdcn_cvt_weight<<<ceil_div_i(tot, 256), 256, 0, stream>>>(
            (const float*)d_in[wd[i].idx], wbf[i], wd[i].dout, wd[i].din, Kp,
            wd[i].transposed);
    }

    // ---- activation buffers ----
    __bf16* x_bf  = (__bf16*)wsAlloc((size_t)N * 2016 * 2);
    __bf16* t1_bf = (__bf16*)wsAlloc((size_t)N * 1024 * 2);
    __bf16* t2_bf = (__bf16*)wsAlloc((size_t)N * 512  * 2);
    __bf16* t3_bf = (__bf16*)wsAlloc((size_t)N * 256  * 2);
    __bf16* z1_bf = (__bf16*)wsAlloc((size_t)N * 128  * 2);
    __bf16* z2_bf = (__bf16*)wsAlloc((size_t)N * 64   * 2);
    __bf16* z3_bf = (__bf16*)wsAlloc((size_t)N * 32   * 2);
    __bf16* d1_bf = (__bf16*)wsAlloc((size_t)N * 256  * 2);
    __bf16* d2_bf = (__bf16*)wsAlloc((size_t)N * 512  * 2);
    __bf16* d3_bf = (__bf16*)wsAlloc((size_t)N * 1024 * 2);
    float*  S1    = (float*)wsAlloc((size_t)N * 2000 * 4);   // GEMM f32 scratch
    float*  GA    = (float*)wsAlloc((size_t)N * 1024 * 4);   // GNN gemm out
    float*  GB    = (float*)wsAlloc((size_t)N * 1024 * 4);   // spmm accumulator
    __bf16* Hc    = (__bf16*)wsAlloc((size_t)N * 1024 * 2);  // GNN combined input
    float*  dmean = (float*)wsAlloc(1024 * 4);
    float*  drstd = (float*)wsAlloc(1024 * 4);

    // ---- helpers ----
    auto gemm = [&](const __bf16* A, const __bf16* Bt, const float* bias, float* C,
                    int M, int Nc, int Kp, int ldc) {
        const int tm2 = M / (16 * GEMM_MT);                       // M % 32 == 0
        const int tn2 = (Nc + 16 * GEMM_NT - 1) / (16 * GEMM_NT);
        const long tiles = (long)tm2 * tn2;
        const int blocks = (int)((tiles + 7) / 8);
        sdcn_gemm_bf16_wmma<<<blocks, 256, 0, stream>>>(A, Bt, bias, C,
                                                        M, Nc, Kp, ldc, tn2, tiles);
    };
    auto bn = [&](float* H, int Nc, int gIdx, int bIdx, int relu,
                  __bf16* outB, float* outF) {
        sdcn_bn_stats<<<Nc, 256, 0, stream>>>(H, N, Nc, dmean, drstd);
        const size_t tot = (size_t)N * Nc;   // all BN dims are multiples of 32
        sdcn_bn_apply<<<ceil_div_i(tot, 256), 256, 0, stream>>>(
            H, dmean, drstd, (const float*)d_in[gIdx], (const float*)d_in[bIdx],
            outF, outB, N, Nc, Nc, relu);
    };

    // ================= Autoencoder =================
    sdcn_cvt_act<<<ceil_div_i((size_t)N * 2016, 256), 256, 0, stream>>>(
        x, x_bf, N, 2000, 2016);

    gemm(x_bf, wbf[0], (const float*)d_in[5], S1, N, 1024, 2016, 1024);
    bn(S1, 1024, 30, 31, 1, t1_bf, nullptr);
    gemm(t1_bf, wbf[1], (const float*)d_in[7], S1, N, 512, 1024, 512);
    bn(S1, 512, 32, 33, 1, t2_bf, nullptr);
    gemm(t2_bf, wbf[2], (const float*)d_in[9], S1, N, 256, 512, 256);
    bn(S1, 256, 34, 35, 1, t3_bf, nullptr);
    gemm(t3_bf, wbf[3], (const float*)d_in[11], S1, N, 128, 256, 128);
    bn(S1, 128, 36, 37, 0, z1_bf, nullptr);
    gemm(z1_bf, wbf[4], (const float*)d_in[13], S1, N, 64, 128, 64);
    bn(S1, 64, 38, 39, 0, z2_bf, nullptr);
    gemm(z2_bf, wbf[5], (const float*)d_in[15], S1, N, 32, 64, 32);
    bn(S1, 32, 40, 41, 0, z3_bf, o_z3);                    // z3 f32 -> d_out
    gemm(z3_bf, wbf[6], (const float*)d_in[17], S1, N, 256, 32, 256);
    bn(S1, 256, 42, 43, 1, d1_bf, nullptr);
    gemm(d1_bf, wbf[7], (const float*)d_in[19], S1, N, 512, 256, 512);
    bn(S1, 512, 44, 45, 1, d2_bf, nullptr);
    gemm(d2_bf, wbf[8], (const float*)d_in[21], S1, N, 1024, 512, 1024);
    bn(S1, 1024, 46, 47, 1, d3_bf, nullptr);

    // x_bar straight into d_out
    gemm(d3_bf, wbf[9], (const float*)d_in[23], o_xbar, N, 2000, 1024, 2000);

    // ================= Decoder heads =================
    const size_t nhead = (size_t)N * 2000;
    gemm(d3_bf, wbf[10], (const float*)d_in[25], S1, N, 2000, 1024, 2000);
    sdcn_heads<<<ceil_div_i(nhead, 256), 256, 0, stream>>>(S1, o_mean, nhead, 0);
    gemm(d3_bf, wbf[11], (const float*)d_in[27], S1, N, 2000, 1024, 2000);
    sdcn_heads<<<ceil_div_i(nhead, 256), 256, 0, stream>>>(S1, o_disp, nhead, 1);
    gemm(d3_bf, wbf[12], (const float*)d_in[29], S1, N, 2000, 1024, 2000);
    sdcn_heads<<<ceil_div_i(nhead, 256), 256, 0, stream>>>(S1, o_pi, nhead, 2);

    // ================= GNN branch =================
    const __bf16* tbufs[6] = {t1_bf, t2_bf, t3_bf, z1_bf, z2_bf, z3_bf};
    const int gout[7] = {1024, 512, 256, 128, 64, 32, 10};
    const __bf16* gin = x_bf;
    int ginKp = 2016;
    for (int L = 0; L < 7; ++L) {
        const int Nc = gout[L];
        gemm(gin, wbf[13 + L], nullptr, GA, N, Nc, ginKp, Nc);
        hipMemsetAsync(GB, 0, (size_t)N * Nc * sizeof(float), stream);
        sdcn_spmm_scatter<<<E, 128, 0, stream>>>(esrc, edst, ew, GA, GB, E, Nc);
        if (L < 6) {
            const size_t tot = (size_t)N * Nc;
            sdcn_combine_relu<<<ceil_div_i(tot, 256), 256, 0, stream>>>(
                GB, tbufs[L], Hc, N, Nc, Nc, SIGMA, 1);
            gin = Hc;
            ginKp = Nc;
        } else {
            sdcn_softmax_rows<<<ceil_div_i((size_t)N, 256), 256, 0, stream>>>(
                GB, o_pred, N, 10);
        }
    }

    // ================= Student-t q =================
    sdcn_student_q<<<ceil_div_i((size_t)N, 256), 256, 0, stream>>>(
        o_z3, cluster, o_q, N);
}
